// MultiHeadAttention_74259984547939
// MI455X (gfx1250) — matrix operand order repaired and checked by reading, not compiled, not run on hardware
//
#include <hip/hip_runtime.h>
#include <hip/hip_bf16.h>

// ---------------------------------------------------------------------------
// Problem constants (from reference): B=2, S=2048, D=1024, H=16, KVH=4, DK=64
// ---------------------------------------------------------------------------
#define BATCH   2
#define SEQ     2048
#define DMODEL  1024
#define NHEAD   16
#define NKVH    4
#define HD      64
#define KVD     (NKVH * HD)      // 256
#define LORA_R  16
#define LORA_SCALE 1.0f
#define MROWS   (BATCH * SEQ)    // 4096 flattened tokens
#define QWAVES  4                // query-tile waves per attention workgroup

typedef _Float16 v8h  __attribute__((ext_vector_type(8)));
typedef _Float16 v16h __attribute__((ext_vector_type(16)));
typedef float    v8f  __attribute__((ext_vector_type(8)));

static __device__ __forceinline__ v16h cat8(v8h lo, v8h hi) {
  return __builtin_shufflevector(lo, hi, 0,1,2,3,4,5,6,7,8,9,10,11,12,13,14,15);
}

#define WMMA_F32_F16(a, b, c) \
  __builtin_amdgcn_wmma_f32_16x16x32_f16(false, (a), false, (b), (short)0, (c), false, false)

// ---------------------------------------------------------------------------
// CDNA5 Tensor Data Mover support (guarded: falls back to manual staging)
// ---------------------------------------------------------------------------
#if __has_builtin(__builtin_amdgcn_tensor_load_to_lds) && __has_builtin(__builtin_amdgcn_s_wait_tensorcnt)
#define HAS_TDM 1
#else
#define HAS_TDM 0
#endif

#if HAS_TDM
typedef unsigned int u32x4 __attribute__((ext_vector_type(4)));
typedef int          i32x4 __attribute__((ext_vector_type(4)));
typedef int          i32x8 __attribute__((ext_vector_type(8)));

// Issue a TDM load of a 2D tile (32 rows x 64 halves) from a row-major
// [MROWS][KVD] fp16 tensor into LDS, padding each 128B row by 16B so the
// LDS row stride is 72 halves (matches the WMMA fragment layout below).
static __device__ __forceinline__ void tdm_load_kv_tile(const _Float16* gtile,
                                                        unsigned lds_off) {
  unsigned long long ga = (unsigned long long)(size_t)gtile;

  u32x4 g0;
  g0[0] = 1u;                                           // count=1 user descriptor
  g0[1] = lds_off;                                      // LDS byte offset
  g0[2] = (unsigned)(ga & 0xFFFFFFFFu);                 // global_addr[31:0]
  g0[3] = (unsigned)((ga >> 32) & 0x1FFFFFFu)           // global_addr[56:32]
          | 0x80000000u;                                // type=2 ("image")

  i32x8 g1;
  // dw0: workgroup_mask=0 | data_size=1 (2B) | pad_enable |
  //      pad_interval=4 (32 DWORDs = one 128B row) | pad_amount=3 (4 DWORDs = 16B)
  g1[0] = (1 << 16) | (1 << 20) | (4 << 22) | (3 << 25);
  g1[1] = (int)(((unsigned)KVD & 0xFFFFu) << 16);       // tensor_dim0[15:0] @63:48
  g1[2] = (int)(((unsigned)KVD >> 16) & 0xFFFFu)        // tensor_dim0[31:16]
          | (int)(((unsigned)MROWS & 0xFFFFu) << 16);   // tensor_dim1[15:0]
  g1[3] = (int)(((unsigned)MROWS >> 16) & 0xFFFFu)      // tensor_dim1[31:16]
          | (64 << 16);                                 // tile_dim0 = 64 elements
  g1[4] = 32;                                           // tile_dim1 = 32, tile_dim2 = 0
  g1[5] = KVD;                                          // tensor_dim0_stride[31:0]
  g1[6] = 0;                                            // stride hi / dim1_stride lo
  g1[7] = 0;

  i32x4 z4 = {0, 0, 0, 0};
#if __clang_major__ >= 23
  i32x8 z8 = {0, 0, 0, 0, 0, 0, 0, 0};
  __builtin_amdgcn_tensor_load_to_lds(g0, g1, z4, z4, z8, 0);
#else
  __builtin_amdgcn_tensor_load_to_lds(g0, g1, z4, z4, 0);
#endif
}
#endif // HAS_TDM

// ---------------------------------------------------------------------------
// Kernel 1: fp32 -> fp16 cast for activations
// ---------------------------------------------------------------------------
__global__ __launch_bounds__(256) void cast_f32_to_f16(const float* __restrict__ in,
                                                       _Float16* __restrict__ out, int n) {
  int i = blockIdx.x * blockDim.x + threadIdx.x;
  if (i < n) out[i] = (_Float16)in[i];
}

// ---------------------------------------------------------------------------
// Kernel 2: fold LoRA into weights: Weff = W + scale * (A @ B), output fp16
// ---------------------------------------------------------------------------
__global__ __launch_bounds__(256) void fuse_lora_weights(const float* __restrict__ W,
                                                         const float* __restrict__ A,
                                                         const float* __restrict__ Bm,
                                                         _Float16* __restrict__ out, int N) {
  int idx = blockIdx.x * blockDim.x + threadIdx.x;   // over D*N elements
  int i = idx / N, j = idx % N;
  float acc = W[idx];
  #pragma unroll
  for (int r = 0; r < LORA_R; ++r)
    acc += LORA_SCALE * A[i * LORA_R + r] * Bm[r * N + j];
  out[idx] = (_Float16)acc;
}

// ---------------------------------------------------------------------------
// Kernel 3: WMMA GEMM  C[MxN] = A[MxK] @ B[KxN]  (A,B fp16, C fp16 or fp32)
// 8 waves, 64x64 block tile, k-step 32; LDS staging with 16B-aligned padded
// rows so A/B fragments are contiguous b128 LDS reads in wave32 WMMA layout.
// ---------------------------------------------------------------------------
template <bool OUT_F32>
__global__ __launch_bounds__(256) void gemm_wmma(const _Float16* __restrict__ A,
                                                 const _Float16* __restrict__ Bw,
                                                 void* __restrict__ Cout,
                                                 int M, int N, int K) {
  __shared__ _Float16 As [64 * 40];   // [row][k]
  __shared__ _Float16 Bts[64 * 40];   // [n][k] transposed

  const int tid  = threadIdx.x;
  const int lane = tid & 31;
  const int w    = tid >> 5;
  const int mw   = w & 3;
  const int nw   = w >> 2;
  const int m0   = blockIdx.y * 64;
  const int n0   = blockIdx.x * 64;
  const int r    = lane & 15;
  const int hh   = lane >> 4;

  const int arow  = tid >> 2, akseg = (tid & 3) * 8;
  const int bkrow = tid >> 3, bnseg = (tid & 7) * 8;

  v8f acc0 = {0.f,0.f,0.f,0.f,0.f,0.f,0.f,0.f};
  v8f acc1 = {0.f,0.f,0.f,0.f,0.f,0.f,0.f,0.f};

  for (int k0 = 0; k0 < K; k0 += 32) {
    const _Float16* agp = A + (size_t)(m0 + arow) * K + k0 + akseg;
    v8h av = *(const v8h*)agp;
    *(v8h*)(As + arow * 40 + akseg) = av;

    const _Float16* bgp = Bw + (size_t)(k0 + bkrow) * N + n0 + bnseg;
    v8h bv = *(const v8h*)bgp;
    #pragma unroll
    for (int j = 0; j < 8; ++j) Bts[(bnseg + j) * 40 + bkrow] = bv[j];

    if (k0 + 32 < K) {
      __builtin_prefetch(agp + 32, 0, 1);
      __builtin_prefetch(bgp + (size_t)32 * N, 0, 1);
    }
    __syncthreads();

    const _Float16* ar = As + (mw * 16 + r) * 40;
    v16h afrag = cat8(*(const v8h*)(ar + hh * 16),
                      *(const v8h*)(ar + hh * 16 + 8));
    {
      const _Float16* br = Bts + (nw * 32 + r) * 40 + hh * 16;
      acc0 = WMMA_F32_F16(afrag, cat8(*(const v8h*)br, *(const v8h*)(br + 8)), acc0);
    }
    {
      const _Float16* br = Bts + (nw * 32 + 16 + r) * 40 + hh * 16;
      acc1 = WMMA_F32_F16(afrag, cat8(*(const v8h*)br, *(const v8h*)(br + 8)), acc1);
    }
    __syncthreads();
  }

  #pragma unroll
  for (int v = 0; v < 8; ++v) {
    int mi = m0 + mw * 16 + hh * 8 + v;
    int ni = n0 + nw * 32 + r;
    if (OUT_F32) {
      ((float*)Cout)[(size_t)mi * N + ni]      = acc0[v];
      ((float*)Cout)[(size_t)mi * N + ni + 16] = acc1[v];
    } else {
      ((_Float16*)Cout)[(size_t)mi * N + ni]      = (_Float16)acc0[v];
      ((_Float16*)Cout)[(size_t)mi * N + ni + 16] = (_Float16)acc1[v];
    }
  }
}

// ---------------------------------------------------------------------------
// Kernel 4: flash attention. 4 waves / workgroup cover 64 queries of one head;
// K/V tiles (32 keys) are staged into LDS once per workgroup — via the Tensor
// Data Mover (double-buffered, TENSORcnt-tracked) when available — and shared
// by all 4 query tiles. QK^T and P@V on WMMA f16->f32, online softmax with
// cross-lane reductions, ALiBi + causal in registers.
// ---------------------------------------------------------------------------
__global__ __launch_bounds__(QWAVES * 32) void attn_wmma(const _Float16* __restrict__ Qp,
                                                         const _Float16* __restrict__ Kp,
                                                         const _Float16* __restrict__ Vp,
                                                         _Float16* __restrict__ Xo) {
  __shared__ _Float16 Kt[2][32 * 72];   // [buf][key][dk], stride 72 halves
  __shared__ _Float16 Vt[2][32 * 72];
  __shared__ _Float16 Pt[QWAVES][16 * 40];

  const int lane = threadIdx.x & 31;
  const int wid  = threadIdx.x >> 5;
  const int r    = lane & 15;
  const int hh   = lane >> 4;
  const int h    = blockIdx.y;
  const int b    = blockIdx.z;
  const int kvh  = h >> 2;                       // GQA: 4 query heads per kv head
  const int q0   = blockIdx.x * 64 + wid * 16;   // this wave's query tile
  const float slope  = exp2f(-(float)(h + 1));   // ALiBi, alpha=1
  const float scale  = 0.125f;                   // 1/sqrt(64)
  const float NEGINF = -3.0e38f;

  // Q A-fragments for the two 32-wide dk steps
  const _Float16* qrow = Qp + (size_t)(b * SEQ + q0 + r) * DMODEL + h * HD;
  v16h aq[2];
  #pragma unroll
  for (int s = 0; s < 2; ++s)
    aq[s] = cat8(*(const v8h*)(qrow + s * 32 + hh * 16),
                 *(const v8h*)(qrow + s * 32 + hh * 16 + 8));

  v8f oacc[4];
  float mrun[8], lrun[8];
  #pragma unroll
  for (int t = 0; t < 4; ++t)
    #pragma unroll
    for (int v = 0; v < 8; ++v) oacc[t][v] = 0.f;
  #pragma unroll
  for (int v = 0; v < 8; ++v) { mrun[v] = NEGINF; lrun[v] = 0.f; }

  const int kbase  = b * SEQ;                       // key row base for this batch
  const int ktiles = (blockIdx.x * 64 + 95) >> 5;   // causal bound for the block

#if HAS_TDM
  if (wid == 0) {   // prologue: issue tile 0 into buffer 0
    tdm_load_kv_tile(Kp + ((size_t)kbase * KVD + kvh * HD),
                     (unsigned)(size_t)&Kt[0][0]);
    tdm_load_kv_tile(Vp + ((size_t)kbase * KVD + kvh * HD),
                     (unsigned)(size_t)&Vt[0][0]);
  }
#endif

  for (int kt = 0; kt < ktiles; ++kt) {
    const int k0 = kt * 32;
#if HAS_TDM
    const int cur = kt & 1;
    if (wid == 0) __builtin_amdgcn_s_wait_tensorcnt(0);
    __syncthreads();                              // tile kt visible to all waves
    if (wid == 0 && kt + 1 < ktiles) {            // issue tile kt+1 into other buffer
      const size_t goff = (size_t)(kbase + k0 + 32) * KVD + kvh * HD;
      tdm_load_kv_tile(Kp + goff, (unsigned)(size_t)&Kt[1 - cur][0]);
      tdm_load_kv_tile(Vp + goff, (unsigned)(size_t)&Vt[1 - cur][0]);
    }
#else
    const int cur = 0;
    {   // manual cooperative staging: 128 threads, 32 keys x 4 segments
      const int key = threadIdx.x >> 2;
      const int seg = (threadIdx.x & 3) * 16;
      const size_t goff = (size_t)(kbase + k0 + key) * KVD + kvh * HD + seg;
      *(v8h*)(Kt[0] + key * 72 + seg)     = *(const v8h*)(Kp + goff);
      *(v8h*)(Kt[0] + key * 72 + seg + 8) = *(const v8h*)(Kp + goff + 8);
      *(v8h*)(Vt[0] + key * 72 + seg)     = *(const v8h*)(Vp + goff);
      *(v8h*)(Vt[0] + key * 72 + seg + 8) = *(const v8h*)(Vp + goff + 8);
    }
    __syncthreads();
#endif

    if (k0 <= q0 + 15) {   // wave-uniform causal early-out (EXEC stays full)
      // ---- scores: S = Q @ K^T, two 16x16 tiles over 32 keys ----
      v8f s0 = {0.f,0.f,0.f,0.f,0.f,0.f,0.f,0.f};
      v8f s1 = {0.f,0.f,0.f,0.f,0.f,0.f,0.f,0.f};
      #pragma unroll
      for (int s = 0; s < 2; ++s) {
        {
          const _Float16* br = Kt[cur] + r * 72 + s * 32 + hh * 16;
          s0 = WMMA_F32_F16(aq[s], cat8(*(const v8h*)br, *(const v8h*)(br + 8)), s0);
        }
        {
          const _Float16* br = Kt[cur] + (16 + r) * 72 + s * 32 + hh * 16;
          s1 = WMMA_F32_F16(aq[s], cat8(*(const v8h*)br, *(const v8h*)(br + 8)), s1);
        }
      }

      // ---- ALiBi + causal + online softmax (row = hh*8+v, cols r / 16+r) ----
      #pragma unroll
      for (int v = 0; v < 8; ++v) {
        const int q  = q0 + hh * 8 + v;
        const int kA = k0 + r;
        const int kB = k0 + 16 + r;
        float sa = s0[v] * scale + slope * (float)(kA - q);
        float sb = s1[v] * scale + slope * (float)(kB - q);
        if (kA > q) sa = NEGINF;
        if (kB > q) sb = NEGINF;

        float t = fmaxf(sa, sb);
        t = fmaxf(t, __shfl_xor(t, 1));
        t = fmaxf(t, __shfl_xor(t, 2));
        t = fmaxf(t, __shfl_xor(t, 4));
        t = fmaxf(t, __shfl_xor(t, 8));
        const float mn = fmaxf(mrun[v], t);

        const float ea = __expf(sa - mn);
        const float eb = __expf(sb - mn);
        float rs = ea + eb;
        rs += __shfl_xor(rs, 1);
        rs += __shfl_xor(rs, 2);
        rs += __shfl_xor(rs, 4);
        rs += __shfl_xor(rs, 8);

        const float alpha = __expf(mrun[v] - mn);
        lrun[v] = lrun[v] * alpha + rs;
        mrun[v] = mn;
        #pragma unroll
        for (int t4 = 0; t4 < 4; ++t4) oacc[t4][v] *= alpha;

        Pt[wid][(hh * 8 + v) * 40 + r]      = (_Float16)ea;
        Pt[wid][(hh * 8 + v) * 40 + 16 + r] = (_Float16)eb;
      }

      // ---- P A-fragment (16x32 over keys); per-wave Pt, same-wave ordering ----
      const _Float16* pr = Pt[wid] + r * 40;
      v16h pf = cat8(*(const v8h*)(pr + hh * 16), *(const v8h*)(pr + hh * 16 + 8));

      // ---- O += P @ V, four 16-wide dk tiles ----
      #pragma unroll
      for (int dt = 0; dt < 4; ++dt) {
        const int n = dt * 16 + r;
        v16h bf;
        #pragma unroll
        for (int i = 0; i < 8; ++i) {
          bf[2 * i]     = Vt[cur][(hh * 16 + 2 * i) * 72 + n];
          bf[2 * i + 1] = Vt[cur][(hh * 16 + 2 * i + 1) * 72 + n];
        }
        oacc[dt] = WMMA_F32_F16(pf, bf, oacc[dt]);
      }
    }
    __syncthreads();   // all reads done before buffer is overwritten
  }

  // ---- normalize and write attention output (fp16 for the O-projection) ----
  #pragma unroll
  for (int v = 0; v < 8; ++v) {
    const float inv = 1.0f / lrun[v];
    _Float16* orow = Xo + (size_t)(b * SEQ + q0 + hh * 8 + v) * DMODEL + h * HD;
    #pragma unroll
    for (int dt = 0; dt < 4; ++dt)
      orow[dt * 16 + r] = (_Float16)(oacc[dt][v] * inv);
  }
}

// ---------------------------------------------------------------------------
// Host orchestration
// ---------------------------------------------------------------------------
extern "C" void kernel_launch(void* const* d_in, const int* in_sizes, int n_in,
                              void* d_out, int out_size, void* d_ws, size_t ws_size,
                              hipStream_t stream) {
  (void)in_sizes; (void)n_in; (void)out_size; (void)ws_size;

  const float* query = (const float*)d_in[0];
  const float* key   = (const float*)d_in[1];
  const float* value = (const float*)d_in[2];
  const float* Wq    = (const float*)d_in[3];
  const float* Wk    = (const float*)d_in[4];
  const float* Wv    = (const float*)d_in[5];
  const float* Wo    = (const float*)d_in[6];
  const float* Aq    = (const float*)d_in[7];
  const float* Bq    = (const float*)d_in[8];
  const float* Ak    = (const float*)d_in[9];
  const float* Bk    = (const float*)d_in[10];
  const float* Av    = (const float*)d_in[11];
  const float* Bv    = (const float*)d_in[12];
  const float* Ao    = (const float*)d_in[13];
  const float* Bo    = (const float*)d_in[14];

  char* ws = (char*)d_ws;
  size_t off = 0;
  auto alloc16 = [&](size_t nelem) -> _Float16* {
    _Float16* p = (_Float16*)(ws + off);
    off += nelem * sizeof(_Float16);
    return p;
  };
  _Float16* q16  = alloc16((size_t)MROWS * DMODEL);
  _Float16* k16  = alloc16((size_t)MROWS * DMODEL);
  _Float16* v16  = alloc16((size_t)MROWS * DMODEL);
  _Float16* wq16 = alloc16((size_t)DMODEL * DMODEL);
  _Float16* wk16 = alloc16((size_t)DMODEL * KVD);
  _Float16* wv16 = alloc16((size_t)DMODEL * KVD);
  _Float16* wo16 = alloc16((size_t)DMODEL * DMODEL);
  _Float16* Qp   = alloc16((size_t)MROWS * DMODEL);
  _Float16* Kp   = alloc16((size_t)MROWS * KVD);
  _Float16* Vp   = alloc16((size_t)MROWS * KVD);
  _Float16* Xa   = alloc16((size_t)MROWS * DMODEL);

  const int nAct = MROWS * DMODEL;
  cast_f32_to_f16<<<nAct / 256, 256, 0, stream>>>(query, q16, nAct);
  cast_f32_to_f16<<<nAct / 256, 256, 0, stream>>>(key,   k16, nAct);
  cast_f32_to_f16<<<nAct / 256, 256, 0, stream>>>(value, v16, nAct);

  fuse_lora_weights<<<(DMODEL * DMODEL) / 256, 256, 0, stream>>>(Wq, Aq, Bq, wq16, DMODEL);
  fuse_lora_weights<<<(DMODEL * KVD)    / 256, 256, 0, stream>>>(Wk, Ak, Bk, wk16, KVD);
  fuse_lora_weights<<<(DMODEL * KVD)    / 256, 256, 0, stream>>>(Wv, Av, Bv, wv16, KVD);
  fuse_lora_weights<<<(DMODEL * DMODEL) / 256, 256, 0, stream>>>(Wo, Ao, Bo, wo16, DMODEL);

  gemm_wmma<false><<<dim3(DMODEL / 64, MROWS / 64), 256, 0, stream>>>(q16, wq16, (void*)Qp, MROWS, DMODEL, DMODEL);
  gemm_wmma<false><<<dim3(KVD    / 64, MROWS / 64), 256, 0, stream>>>(k16, wk16, (void*)Kp, MROWS, KVD,    DMODEL);
  gemm_wmma<false><<<dim3(KVD    / 64, MROWS / 64), 256, 0, stream>>>(v16, wv16, (void*)Vp, MROWS, KVD,    DMODEL);

  attn_wmma<<<dim3(SEQ / 64, NHEAD, BATCH), QWAVES * 32, 0, stream>>>(Qp, Kp, Vp, Xa);

  gemm_wmma<true><<<dim3(DMODEL / 64, MROWS / 64), 256, 0, stream>>>(Xa, wo16, d_out, MROWS, DMODEL, DMODEL);
}
